// MultiHeadAttention_45440753992045
// MI455X (gfx1250) — matrix operand order repaired and checked by reading, not compiled, not run on hardware
//
#include <hip/hip_runtime.h>
#include <stdint.h>

// ---------- types ----------
typedef __attribute__((ext_vector_type(16))) __bf16       v16bf;
typedef __attribute__((ext_vector_type(8)))  float        v8f;
typedef __attribute__((ext_vector_type(4)))  unsigned int u32x4;

union FragU { u32x4 q[2]; v16bf v; };

__device__ __forceinline__ unsigned short f32_to_bf16(float f) {
  unsigned int u = __float_as_uint(f);
  u += 0x7FFFu + ((u >> 16) & 1u);   // round-to-nearest-even
  return (unsigned short)(u >> 16);
}

// A fragment (16x32 bf16, M x K), src row-major [rows x ld], tile origin at src.
// Layout (ISA 7.12.2): lanes 0-15: row=lane, v0..3 K=0..7, v4..7 K=16..23;
//                      lanes 16-31: row=lane-16, v0..3 K=8..15, v4..7 K=24..31.
__device__ __forceinline__ v16bf load_frag_a(const unsigned short* src, int ld, int lane) {
  int row  = lane & 15;
  int koff = (lane >> 4) << 3;          // 0 or 8
  const unsigned short* p = src + row * ld + koff;
  FragU f;
  f.q[0] = *reinterpret_cast<const u32x4*>(p);
  f.q[1] = *reinterpret_cast<const u32x4*>(p + 16);
  return f.v;
}

// B fragment (32x16 bf16, K x N), src is N-MAJOR: src[n*ld + k] == B[k][n].
// Layout: lanes 0-15: n=lane, v0..3 K=0..7, v4..7 K=16..23; lanes 16-31: n=lane-16, v0..3 K=8..15, v4..7 K=24..31.
__device__ __forceinline__ v16bf load_frag_b(const unsigned short* src, int ld, int lane) {
  int n    = lane & 15;
  int koff = (lane >> 4) << 3;          // 0 or 8
  const unsigned short* p = src + n * ld + koff;
  FragU f;
  f.q[0] = *reinterpret_cast<const u32x4*>(p);
  f.q[1] = *reinterpret_cast<const u32x4*>(p + 16);
  return f.v;
}

__device__ __forceinline__ v8f wmma_bf16(v16bf a, v16bf b, v8f c) {
  return __builtin_amdgcn_wmma_f32_16x16x32_bf16(false, a, false, b, (short)0, c, false, false);
}

// ---------- stage 0: converts ----------
__global__ void convert_f32_bf16_kernel(const float* __restrict__ in,
                                        unsigned short* __restrict__ out, int n) {
  int i = blockIdx.x * 256 + threadIdx.x;
  if (i < n) out[i] = f32_to_bf16(in[i]);
}

// W [K x N] row-major fp32  ->  Wt [N x K] bf16 (n-major for B fragments)
__global__ void transpose_f32_bf16_kernel(const float* __restrict__ in,
                                          unsigned short* __restrict__ out, int K, int N) {
  int i = blockIdx.x * 256 + threadIdx.x;
  if (i < K * N) {
    int k = i / N, n = i % N;
    out[n * K + k] = f32_to_bf16(in[i]);
  }
}

// ---------- stage 1: QKV projection (wave = 32x32 tile, 2x2 register blocking) ----------
// Xb [8192 x 1024] bf16, Wqt [3072 x 1024] bf16 (n-major).
// col c -> head h=c/192, r=c%192: r<64 -> Q, r<128 -> K, else -> Vt (d-major).
__global__ __launch_bounds__(256) void qkv_gemm_kernel(
    const unsigned short* __restrict__ Xb, const unsigned short* __restrict__ Wqt,
    const float* __restrict__ bqkv,
    unsigned short* __restrict__ Qb, unsigned short* __restrict__ Kb,
    unsigned short* __restrict__ Vt)
{
  const int NT = 96;                          // 3072/32 column tiles
  int gwave = (blockIdx.x * 256 + (int)threadIdx.x) >> 5;
  int tn = gwave % NT, tm = gwave / NT;
  int lane = threadIdx.x & 31;

  v8f acc[2][2] = {{{}, {}}, {{}, {}}};
  const unsigned short* arow0 = Xb  + (tm * 32 +  0) * 1024;
  const unsigned short* arow1 = Xb  + (tm * 32 + 16) * 1024;
  const unsigned short* brow0 = Wqt + (tn * 32 +  0) * 1024;
  const unsigned short* brow1 = Wqt + (tn * 32 + 16) * 1024;
  for (int k0 = 0; k0 < 1024; k0 += 32) {
    v16bf a0 = load_frag_a(arow0 + k0, 1024, lane);
    v16bf a1 = load_frag_a(arow1 + k0, 1024, lane);
    v16bf b0 = load_frag_b(brow0 + k0, 1024, lane);
    v16bf b1 = load_frag_b(brow1 + k0, 1024, lane);
    acc[0][0] = wmma_bf16(a0, b0, acc[0][0]);
    acc[0][1] = wmma_bf16(a0, b1, acc[0][1]);
    acc[1][0] = wmma_bf16(a1, b0, acc[1][0]);
    acc[1][1] = wmma_bf16(a1, b1, acc[1][1]);
  }

  int lane15 = lane & 15;
  int half8 = (lane >> 4) << 3;
#pragma unroll
  for (int ni = 0; ni < 2; ++ni) {
    int col = tn * 32 + ni * 16 + lane15;
    float bv = bqkv[col];
    int h = col / 192, r = col % 192;
#pragma unroll
    for (int mi = 0; mi < 2; ++mi) {
#pragma unroll
      for (int rr = 0; rr < 8; ++rr) {
        int row = tm * 32 + mi * 16 + rr + half8;  // global token row in [0,8192)
        int bidx = row >> 11, s = row & 2047;
        int bh = bidx * 16 + h;
        unsigned short val = f32_to_bf16(acc[mi][ni][rr] + bv);
        if (r < 64)       Qb[(bh * 2048 + s) * 64 + r]          = val;
        else if (r < 128) Kb[(bh * 2048 + s) * 64 + (r - 64)]   = val;
        else              Vt[(bh * 64 + (r - 128)) * 2048 + s]  = val;   // transposed
      }
    }
  }
}

// ---------- stage 2: causal flash attention ----------
// Per wave: 16 queries x full hd=64. Key blocks of 32 (two 16-col score tiles).
__global__ __launch_bounds__(256) void attn_kernel(
    const unsigned short* __restrict__ Qb, const unsigned short* __restrict__ Kb,
    const unsigned short* __restrict__ Vt, unsigned short* __restrict__ Ob)
{
  __shared__ __align__(16) unsigned short lds[8 * 512];   // 16x32 bf16 P-tile per wave
  int wid = threadIdx.x >> 5, lane = threadIdx.x & 31;
  int bh = blockIdx.x >> 4, qchunk = blockIdx.x & 15;
  int q0 = qchunk * 128 + wid * 16;

  const unsigned short* Q = Qb + bh * (2048 * 64);
  const unsigned short* K = Kb + bh * (2048 * 64);
  const unsigned short* V = Vt + bh * (64 * 2048);
  unsigned short*       O = Ob + bh * (2048 * 64);
  unsigned short*    plds = lds + wid * 512;

  int half = lane >> 4, ln = lane & 15, half8 = half << 3;

  v16bf qa0 = load_frag_a(Q + q0 * 64,      64, lane);    // d = 0..31
  v16bf qa1 = load_frag_a(Q + q0 * 64 + 32, 64, lane);    // d = 32..63

  float m[8], l[8];
  v8f o0 = {}, o1 = {}, o2 = {}, o3 = {};
#pragma unroll
  for (int r = 0; r < 8; ++r) { m[r] = -1e30f; l[r] = 0.f; }

  const float scale = 0.125f;                 // 1/sqrt(64)
  int kend = q0 + 16;                         // causal bound for this wave
  for (int kb = 0; kb < kend; kb += 32) {
    v8f s0 = {}, s1 = {};
    s0 = wmma_bf16(qa0, load_frag_b(K + kb * 64,            64, lane), s0);
    s0 = wmma_bf16(qa1, load_frag_b(K + kb * 64 + 32,       64, lane), s0);
    s1 = wmma_bf16(qa0, load_frag_b(K + (kb + 16) * 64,     64, lane), s1);
    s1 = wmma_bf16(qa1, load_frag_b(K + (kb + 16) * 64 + 32,64, lane), s1);

    float alpha[8];
#pragma unroll
    for (int r = 0; r < 8; ++r) {
      int qi = q0 + r + half8;
      float v0 = (kb + ln      <= qi) ? s0[r] * scale : -1e30f;
      float v1 = (kb + 16 + ln <= qi) ? s1[r] * scale : -1e30f;
      float mx = fmaxf(v0, v1);               // row reduce over 16-lane half
      mx = fmaxf(mx, __shfl_xor(mx, 1));
      mx = fmaxf(mx, __shfl_xor(mx, 2));
      mx = fmaxf(mx, __shfl_xor(mx, 4));
      mx = fmaxf(mx, __shfl_xor(mx, 8));
      float mnew = fmaxf(m[r], mx);
      float a  = __expf(m[r] - mnew);
      float p0 = __expf(v0 - mnew);
      float p1 = __expf(v1 - mnew);
      float rs = p0 + p1;
      rs += __shfl_xor(rs, 1);
      rs += __shfl_xor(rs, 2);
      rs += __shfl_xor(rs, 4);
      rs += __shfl_xor(rs, 8);
      l[r] = l[r] * a + rs;
      m[r] = mnew;
      alpha[r] = a;
      // C/D layout -> LDS (row-major 16x32 bf16) for A-layout reload
      plds[(r + half8) * 32 + ln]      = f32_to_bf16(p0);
      plds[(r + half8) * 32 + 16 + ln] = f32_to_bf16(p1);
    }
    asm volatile("s_wait_dscnt 0" ::: "memory");   // cross-lane LDS dependency (same wave)
    v16bf pa = load_frag_a(plds, 32, lane);
#pragma unroll
    for (int r = 0; r < 8; ++r) {
      o0[r] *= alpha[r]; o1[r] *= alpha[r]; o2[r] *= alpha[r]; o3[r] *= alpha[r];
    }
    o0 = wmma_bf16(pa, load_frag_b(V +  0 * 2048 + kb, 2048, lane), o0);
    o1 = wmma_bf16(pa, load_frag_b(V + 16 * 2048 + kb, 2048, lane), o1);
    o2 = wmma_bf16(pa, load_frag_b(V + 32 * 2048 + kb, 2048, lane), o2);
    o3 = wmma_bf16(pa, load_frag_b(V + 48 * 2048 + kb, 2048, lane), o3);
  }

#pragma unroll
  for (int r = 0; r < 8; ++r) {
    float inv = 1.0f / l[r];
    int row = q0 + r + half8;
    O[row * 64 + ln]      = f32_to_bf16(o0[r] * inv);
    O[row * 64 + 16 + ln] = f32_to_bf16(o1[r] * inv);
    O[row * 64 + 32 + ln] = f32_to_bf16(o2[r] * inv);
    O[row * 64 + 48 + ln] = f32_to_bf16(o3[r] * inv);
  }
}

// ---------- stage 3: output projection (wave = 32x32 tile, 2x2 register blocking) ----------
// Ab = O viewed as [8192 x 1024] bf16 (faithful to reference's scramble-free reshape),
// Wot [1024 x 1024] bf16 (n-major), out fp32 + bias.
__global__ __launch_bounds__(256) void out_gemm_kernel(
    const unsigned short* __restrict__ Ab, const unsigned short* __restrict__ Wot,
    const float* __restrict__ bo, float* __restrict__ out)
{
  const int NT = 32;                          // 1024/32 column tiles
  int gwave = (blockIdx.x * 256 + (int)threadIdx.x) >> 5;
  int tn = gwave % NT, tm = gwave / NT;
  int lane = threadIdx.x & 31;

  v8f acc[2][2] = {{{}, {}}, {{}, {}}};
  const unsigned short* arow0 = Ab  + (tm * 32 +  0) * 1024;
  const unsigned short* arow1 = Ab  + (tm * 32 + 16) * 1024;
  const unsigned short* brow0 = Wot + (tn * 32 +  0) * 1024;
  const unsigned short* brow1 = Wot + (tn * 32 + 16) * 1024;
  for (int k0 = 0; k0 < 1024; k0 += 32) {
    v16bf a0 = load_frag_a(arow0 + k0, 1024, lane);
    v16bf a1 = load_frag_a(arow1 + k0, 1024, lane);
    v16bf b0 = load_frag_b(brow0 + k0, 1024, lane);
    v16bf b1 = load_frag_b(brow1 + k0, 1024, lane);
    acc[0][0] = wmma_bf16(a0, b0, acc[0][0]);
    acc[0][1] = wmma_bf16(a0, b1, acc[0][1]);
    acc[1][0] = wmma_bf16(a1, b0, acc[1][0]);
    acc[1][1] = wmma_bf16(a1, b1, acc[1][1]);
  }

  int lane15 = lane & 15;
  int half8 = (lane >> 4) << 3;
#pragma unroll
  for (int ni = 0; ni < 2; ++ni) {
    int col = tn * 32 + ni * 16 + lane15;
    float bv = bo[col];
#pragma unroll
    for (int mi = 0; mi < 2; ++mi) {
#pragma unroll
      for (int rr = 0; rr < 8; ++rr) {
        int row = tm * 32 + mi * 16 + rr + half8;
        out[row * 1024 + col] = acc[mi][ni][rr] + bv;
      }
    }
  }
}

// ---------- launch ----------
extern "C" void kernel_launch(void* const* d_in, const int* in_sizes, int n_in,
                              void* d_out, int out_size, void* d_ws, size_t ws_size,
                              hipStream_t stream) {
  const float* x     = (const float*)d_in[0];   // [4,2048,1024]
  const float* W_qkv = (const float*)d_in[1];   // [1024,3072]
  const float* b_qkv = (const float*)d_in[2];   // [3072]
  const float* W_o   = (const float*)d_in[3];   // [1024,1024]
  const float* b_o   = (const float*)d_in[4];   // [1024]
  float* out = (float*)d_out;                   // [4,2048,1024] fp32

  char* ws = (char*)d_ws;
  size_t off = 0;
  auto alloc = [&](size_t bytes) -> void* {
    void* p = ws + off; off += (bytes + 255) & ~(size_t)255; return p;
  };
  unsigned short* Xb  = (unsigned short*)alloc((size_t)8192 * 1024 * 2);  // reused as Ob
  unsigned short* Wqt = (unsigned short*)alloc((size_t)3072 * 1024 * 2);
  unsigned short* Wot = (unsigned short*)alloc((size_t)1024 * 1024 * 2);
  unsigned short* Qb  = (unsigned short*)alloc((size_t)64 * 2048 * 64 * 2);
  unsigned short* Kb  = (unsigned short*)alloc((size_t)64 * 2048 * 64 * 2);
  unsigned short* Vt  = (unsigned short*)alloc((size_t)64 * 64 * 2048 * 2);
  unsigned short* Ob  = Xb;   // Xb fully consumed by qkv_gemm before attn writes Ob

  convert_f32_bf16_kernel<<<(8192 * 1024 + 255) / 256, 256, 0, stream>>>(x, Xb, 8192 * 1024);
  transpose_f32_bf16_kernel<<<(1024 * 3072 + 255) / 256, 256, 0, stream>>>(W_qkv, Wqt, 1024, 3072);
  transpose_f32_bf16_kernel<<<(1024 * 1024 + 255) / 256, 256, 0, stream>>>(W_o, Wot, 1024, 1024);

  // 256 x 96 tiles of 32x32, 8 waves/block
  qkv_gemm_kernel<<<256 * 96 / 8, 256, 0, stream>>>(Xb, Wqt, b_qkv, Qb, Kb, Vt);

  // 64 (b,h) x 16 query-chunks of 128
  attn_kernel<<<64 * 16, 256, 0, stream>>>(Qb, Kb, Vt, Ob);

  // 256 x 32 tiles of 32x32
  out_gemm_kernel<<<256 * 32 / 8, 256, 0, stream>>>(Ob, Wot, b_o, out);
}
